// E3NN_Model_52510270161536
// MI455X (gfx1250) — matrix operand order repaired and checked by reading, not compiled, not run on hardware
//
#include <hip/hip_runtime.h>
#include <math.h>
#include <stdint.h>

// ---------------------------------------------------------------------------
// e3nn GNN forward on MI455X (gfx1250, wave32).
//
// Only layer-2 paths with o==0 reach the output (outs[1], outs[2] are dead in
// the reference), so we run 6 conv paths, each of the form
//   msg = (R(d) @ z) * t,  R = (relu(rbf@W1+b1) @ W2 + b2)  [16x16 per edge]
// The h@W2 (12x256) product is batched over 16 edges per wave with
// v_wmma_f32_16x16x32_f16 (16 N-tiles, B-tiles pinned in VGPRs across 8 edge
// groups per block), R staged through padded LDS for the R@z matvec.
// Node-side linears (in_lin0, out_lin, in_lin, out_lin2) are also WMMA.
// The z gather uses GLOBAL_LOAD_ASYNC_TO_LDS_B128 + s_wait_asynccnt, and the
// next edge group is prefetched with global_prefetch_b8.
// Scatter-adds are native f32 global atomics (accumulators ~40MB << 192MB L2).
// ---------------------------------------------------------------------------

typedef __attribute__((ext_vector_type(16))) _Float16 v16h;
typedef __attribute__((ext_vector_type(8)))  float    v8f;

#define N_NODES 50000
#define N_EDGES 800000
#define NBATCH  64
#define EG      8          // edge groups (of 16 edges) per block

__device__ __forceinline__ v8f wmma16(v16h a, v16h b, v8f c) {
  // D = A(16xK) * B(Kx16) + C, f16 in / f32 out
  return __builtin_amdgcn_wmma_f32_16x16x32_f16(false, a, false, b, (short)0, c,
                                                false, false);
}

// async global -> LDS copy, 16B per lane; tracked by ASYNCcnt
__device__ __forceinline__ void async_copy_b128(uint32_t lds_off, const void* g) {
  asm volatile("global_load_async_to_lds_b128 %0, %1, off"
               :: "v"(lds_off), "v"((unsigned long long)(uintptr_t)g)
               : "memory");
}
__device__ __forceinline__ void wait_async() {
  asm volatile("s_wait_asynccnt 0x0" ::: "memory");
}

// shifted softplus: log(1+e^x) - log 2 (numerically stable)
__device__ __forceinline__ float sspf(float x) {
  return fmaxf(x, 0.f) + log1pf(expf(-fabsf(x))) - 0.69314718055994531f;
}

// A-operand (16xK f16, K padded to 32). Row-major source, row length `stride`.
// lanes 0-15: row M=lane, elems 0-15 = K(kbase+0..15)
// lanes 16-31: row M=lane-16, elems 0-15 = K(kbase+16..31)
__device__ __forceinline__ v16h packA(const float* __restrict__ base, int stride,
                                      int row, int kbase, int kmax, int lane) {
  v16h a;
  const int off = kbase + ((lane & 16) ? 16 : 0);
#pragma unroll
  for (int j = 0; j < 8; ++j) {
    int k  = off + j;
    int k2 = off + 8 + j;
    a[j]     = (k  < kmax) ? (_Float16)base[row * stride + k ] : (_Float16)0.f;
    a[8 + j] = (k2 < kmax) ? (_Float16)base[row * stride + k2] : (_Float16)0.f;
  }
  return a;
}

// B-operand (Kx16 f16): lanes 0-15 col N=lane hold K=0..15; lane+16 K=16..31.
// Source addressed as base[k*ldk + col].
__device__ __forceinline__ v16h packB(const float* __restrict__ base, int ldk,
                                      int col, int kbase, int kmax, int lane) {
  v16h b;
  const int off = kbase + ((lane & 16) ? 16 : 0);
#pragma unroll
  for (int j = 0; j < 16; ++j) {
    int k = off + j;
    b[j] = (k < kmax) ? (_Float16)base[k * ldk + col] : (_Float16)0.f;
  }
  return b;
}

// B-operand where B[k][col] = M[col*ld + k] (i.e. B = M^T, M row-major)
__device__ __forceinline__ v16h packBT(const float* __restrict__ base, int ld,
                                       int col, int kbase, int kmax, int lane) {
  v16h b;
  const int off = kbase + ((lane & 16) ? 16 : 0);
#pragma unroll
  for (int j = 0; j < 16; ++j) {
    int k = off + j;
    b[j] = (k < kmax) ? (_Float16)base[col * ld + k] : (_Float16)0.f;
  }
  return b;
}

// ---------------------------------------------------------------------------
__global__ void k_zero(float* __restrict__ p, int n) {
  int i = blockIdx.x * blockDim.x + threadIdx.x;
  if (i < n) p[i] = 0.f;
}

// h0[n][u] = sum_v W[u][v] x[n][v] / sqrt(32);  one wave per 16-node tile.
__global__ void k_node_h0(const float* __restrict__ x,
                          const float* __restrict__ W,
                          float* __restrict__ h0) {
  const int lane = threadIdx.x;
  const int n0   = blockIdx.x * 16;
  const int me   = lane & 15;
  const int half = lane >> 4;
  v16h a = packA(x, 32, n0 + me, 0, 32, lane);   // A: 16 nodes x 32 inputs
  v16h b = packBT(W, 32, me, 0, 32, lane);       // B: 32 x 16   (= W^T)
  v8f  c = {0.f,0.f,0.f,0.f,0.f,0.f,0.f,0.f};
  v8f  d = wmma16(a, b, c);
  const float sc = 0.17677669529663687f;         // 1/sqrt(32)
#pragma unroll
  for (int g = 0; g < 8; ++g) {
    int n = n0 + g + half * 8;
    h0[n * 16 + me] = d[g] * sc;
  }
}

// ---------------------------------------------------------------------------
struct Radial { const float *W1, *b1, *W2, *b2; };  // (12x12), 12, (12x256), 256
struct EdgeArgs {
  const int*   ei;          // (2, E)
  const float* ea;          // (E, 3)
  const float* inp[3];      // per-path input feature array
  float*       outp[3];     // per-path accumulator
  Radial       rad[3];
};

// One wave per block; EG groups of 16 edges per block.
// LAYER==1: paths (0,o,o), o=0..2, z = h0[src], scatter r ⊗ Y_o/sqrt(2o+1).
// LAYER==2: paths (l,l,0), z = <ins_l, Y_l>/sqrt(2l+1), scatter r into slot l.
template <int LAYER>
__global__ void k_edge(EdgeArgs A) {
  __shared__ float h_lds[16][12];
  __shared__ float z_lds[16][20];    // padded (20) to spread LDS banks
  __shared__ float R_lds[16][260];   // padded (260): 4-bank stride per edge

  const int lane = threadIdx.x;
  const int me   = lane & 15;
  const int half = lane >> 4;

#pragma unroll
  for (int p = 0; p < 3; ++p) {
    const Radial rp = A.rad[p];
    const float invsp = (p == 0) ? 1.f
                      : (p == 1) ? 0.57735026918962576f : 0.44721359549995794f;

    // ---- W2/b2 WMMA B-tiles: built once per block, pinned in VGPRs --------
    v16h  Bt[16];
    float biasv[16];
#pragma unroll
    for (int t = 0; t < 16; ++t) {
      Bt[t]    = packB(rp.W2, 256, t * 16 + me, 0, 12, lane);
      biasv[t] = rp.b2[t * 16 + me];
    }

#pragma unroll 1
    for (int g = 0; g < EG; ++g) {
      const int e = (blockIdx.x * EG + g) * 16 + me;

      // prefetch the next group's edge data (global_prefetch_b8)
      if (g + 1 < EG) {
        const int en = e + 16;
        __builtin_prefetch(&A.ea[en * 3], 0, 3);
        __builtin_prefetch(&A.ei[en], 0, 3);
        __builtin_prefetch(&A.ei[N_EDGES + en], 0, 3);
      }

      const int src = A.ei[e];
      const int dst = A.ei[N_EDGES + e];

      // geometry (computed redundantly by both halves of the lane pair)
      const float ex = A.ea[e * 3 + 0], ey = A.ea[e * 3 + 1], ez = A.ea[e * 3 + 2];
      const float d  = sqrtf(ex * ex + ey * ey + ez * ez + 1e-8f);
      const float iv = 1.f / d;
      const float ux = ex * iv, uy = ey * iv, uz = ez * iv;
      const float s3 = 1.7320508075688772f, s15 = 3.872983346207417f,
                  s5 = 2.23606797749979f;
      float Y[5];
      if (p == 1) {
        Y[0] = s3 * uy; Y[1] = s3 * uz; Y[2] = s3 * ux; Y[3] = 0.f; Y[4] = 0.f;
      } else {
        Y[0] = s15 * ux * uy;
        Y[1] = s15 * uy * uz;
        Y[2] = 0.5f * s5 * (3.f * uz * uz - 1.f);
        Y[3] = s15 * ux * uz;
        Y[4] = 0.5f * s15 * (ux * ux - uy * uy);
      }

      // radial basis (12 gaussians on [0,5])
      float basis[12];
      const float isig = 11.f / 5.f;  // 1/sig, sig = 5/11
#pragma unroll
      for (int k = 0; k < 12; ++k) {
        float mu = 5.f * (float)k / 11.f;
        float t  = (d - mu) * isig;
        basis[k] = expf(-0.5f * t * t);
      }

      // ---- h = relu(basis @ W1 + b1): each lane computes 6 of 12
      {
        const int j0 = half * 6;
#pragma unroll
        for (int j = 0; j < 6; ++j) {
          int jj = j0 + j;
          float acc = rp.b1[jj];
#pragma unroll
          for (int q = 0; q < 12; ++q)
            acc = fmaf(basis[q], rp.W1[q * 12 + jj], acc);
          h_lds[me][jj] = fmaxf(acc, 0.f);
        }
      }

      // ---- z vector (16): async copy for plain gathers, VALU for Y-contraction
      {
        const int v0 = half * 8;
        if (LAYER == 1 || p == 0) {
          const float* zsrc = A.inp[p] + src * 16 + v0;          // 16B aligned
          uint32_t zoff = (uint32_t)(uintptr_t)&z_lds[me][v0];
          async_copy_b128(zoff,      zsrc);
          async_copy_b128(zoff + 16, zsrc + 4);
        } else {
          const int ml = 2 * p + 1;
#pragma unroll
          for (int v = 0; v < 8; ++v) {
            float acc = 0.f;
            for (int ii = 0; ii < ml; ++ii)
              acc = fmaf(A.inp[p][(src * 16 + v0 + v) * ml + ii], Y[ii], acc);
            z_lds[me][v0 + v] = acc * invsp;
          }
        }
      }
      wait_async();
      __syncthreads();

      // ---- A operand: 16 edges x 12 (K padded to 32)
      v16h a;
      {
        const int k0 = half * 16;
#pragma unroll
        for (int j = 0; j < 8; ++j) {
          int k = k0 + j;
          a[j]     = (k < 12) ? (_Float16)h_lds[me][k] : (_Float16)0.f;
          int k2 = k0 + 8 + j;
          a[8 + j] = (k2 < 12) ? (_Float16)h_lds[me][k2] : (_Float16)0.f;
        }
      }

      // ---- R = h @ W2 + b2 : 16 WMMA tiles (tile t == output row u)
#pragma unroll
      for (int t = 0; t < 16; ++t) {
        const float b0 = biasv[t];
        v8f c = {b0, b0, b0, b0, b0, b0, b0, b0};
        v8f dd = wmma16(a, Bt[t], c);
#pragma unroll
        for (int gg = 0; gg < 8; ++gg) {
          int er = gg + half * 8;                // edge row of this D fragment
          R_lds[er][t * 16 + me] = dd[gg];       // R[e][u=t][v=me]
        }
      }
      __syncthreads();

      // ---- r[u] = sum_v R[u][v] z[v] ; lane pair splits u-range
      float zr[16];
#pragma unroll
      for (int q = 0; q < 4; ++q) {
        float4 zz = *reinterpret_cast<const float4*>(&z_lds[me][q * 4]);
        zr[q * 4 + 0] = zz.x; zr[q * 4 + 1] = zz.y;
        zr[q * 4 + 2] = zz.z; zr[q * 4 + 3] = zz.w;
      }
      float r[8];
#pragma unroll
      for (int uu = 0; uu < 8; ++uu) {
        const int u = half * 8 + uu;
        float acc = 0.f;
#pragma unroll
        for (int q = 0; q < 4; ++q) {
          float4 rv = *reinterpret_cast<const float4*>(&R_lds[me][u * 16 + q * 4]);
          acc = fmaf(rv.x, zr[q * 4 + 0], acc);
          acc = fmaf(rv.y, zr[q * 4 + 1], acc);
          acc = fmaf(rv.z, zr[q * 4 + 2], acc);
          acc = fmaf(rv.w, zr[q * 4 + 3], acc);
        }
        r[uu] = acc * 0.25f;   // /sqrt(F)
      }

      // ---- scatter-add to destination node accumulator
      if (LAYER == 1) {
        const int mo = 2 * p + 1;
        float tv[5];
        tv[0] = (p == 0) ? 1.f : Y[0] * invsp;
        if (p > 0) {
          for (int m = 1; m < 5; ++m) tv[m] = Y[m] * invsp;
        }
#pragma unroll
        for (int uu = 0; uu < 8; ++uu) {
          const int u = half * 8 + uu;
          for (int m = 0; m < mo; ++m)
            unsafeAtomicAdd(&A.outp[p][(dst * 16 + u) * mo + m], r[uu] * tv[m]);
        }
      } else {
#pragma unroll
        for (int uu = 0; uu < 8; ++uu) {
          const int u = half * 8 + uu;
          unsafeAtomicAdd(&A.outp[0][dst * 48 + p * 16 + u], r[uu]);
        }
      }
      __syncthreads();   // LDS reused by next group
    }
  }
}

// ---------------------------------------------------------------------------
// Layer-1 node post: norm -> out_lin (WMMA) -> nonlin -> layer2 in_lin (WMMA)
struct L1PostArgs {
  const float* acc[3];
  const float* outW[3];   // (16,16)
  const float* bias[3];   // (16)
  const float* inW2[3];   // (16,16)
  float*       ins[3];
};
__global__ void k_l1post(L1PostArgs A) {
  __shared__ float fshare[16][16][5];
  __shared__ float sc[16];
  const int lane = threadIdx.x;
  const int me   = lane & 15;
  const int half = lane >> 4;
  const int n0   = blockIdx.x * 16;

  for (int o = 0; o < 3; ++o) {
    const int mo = 2 * o + 1;
    const float* acc = A.acc[o];
    if (half == 0) {                       // per-node RMS scale
      float ss = 0.f;
      for (int q = 0; q < 16 * mo; ++q) {
        float v = acc[(n0 + me) * 16 * mo + q];
        ss = fmaf(v, v, ss);
      }
      sc[me] = rsqrtf(ss / (float)(16 * mo) + 1e-8f) * 0.25f;  // fold /sqrt(16)
    }
    __syncthreads();

    v16h aW = packA(A.outW[o], 16, me, 0, 16, lane);
    for (int t = 0; t < mo; ++t) {
      const int ci = t * 16 + me;          // column = (node, m) pair
      const int nl = ci / mo, m = ci - nl * mo;
      v16h b;
      const int off = (lane & 16) ? 16 : 0;
#pragma unroll
      for (int j = 0; j < 16; ++j) {
        int v = off + j;
        b[j] = (v < 16) ? (_Float16)(acc[((n0 + nl) * 16 + v) * mo + m] * sc[nl])
                        : (_Float16)0.f;
      }
      v8f c = {0.f,0.f,0.f,0.f,0.f,0.f,0.f,0.f};
      v8f dd = wmma16(aW, b, c);
#pragma unroll
      for (int g = 0; g < 8; ++g) fshare[nl][g + half * 8][m] = dd[g];
    }
    __syncthreads();

    // gate nonlinearity on 256 (n,u) pairs: 8 per lane
#pragma unroll
    for (int pp = 0; pp < 8; ++pp) {
      const int pr = lane * 8 + pp;
      const int nl = pr >> 4, u = pr & 15;
      if (o == 0) {
        fshare[nl][u][0] = sspf(fshare[nl][u][0]);
      } else {
        float ssum = 1e-8f;
        for (int m = 0; m < mo; ++m) {
          float v = fshare[nl][u][m];
          ssum = fmaf(v, v, ssum);
        }
        float nm  = sqrtf(ssum);
        float fac = sspf(nm + A.bias[o][u]) / nm;
        for (int m = 0; m < mo; ++m) fshare[nl][u][m] *= fac;
      }
    }
    __syncthreads();

    // layer-2 in_lin (fused)
    v16h aW2 = packA(A.inW2[o], 16, me, 0, 16, lane);
    for (int t = 0; t < mo; ++t) {
      const int ci = t * 16 + me;
      const int nl = ci / mo, m = ci - nl * mo;
      v16h b;
      const int off = (lane & 16) ? 16 : 0;
#pragma unroll
      for (int j = 0; j < 16; ++j) {
        int v = off + j;
        b[j] = (v < 16) ? (_Float16)(fshare[nl][v][m] * 0.25f) : (_Float16)0.f;
      }
      v8f c = {0.f,0.f,0.f,0.f,0.f,0.f,0.f,0.f};
      v8f dd = wmma16(aW2, b, c);
#pragma unroll
      for (int g = 0; g < 8; ++g) {
        int u = g + half * 8;
        A.ins[o][((n0 + nl) * 16 + u) * mo + m] = dd[g];
      }
    }
    __syncthreads();
  }
}

// ---------------------------------------------------------------------------
// Layer-2 node post (o==0 only): norm -> out_lin0 (16x48, 2 K-chunk WMMA) ->
// ssp -> batch pooling (atomics).
struct L2PostArgs {
  const float* acc2;   // (N, 48)
  const float* W0;     // (16, 48)
  const int*   batch;
  float*       pool;   // (64, 16)
  float*       cnt;    // (64)
};
__global__ void k_l2post(L2PostArgs A) {
  __shared__ float sc[16];
  const int lane = threadIdx.x;
  const int me   = lane & 15;
  const int half = lane >> 4;
  const int n0   = blockIdx.x * 16;

  if (half == 0) {
    float ss = 0.f;
    for (int q = 0; q < 48; ++q) {
      float v = A.acc2[(n0 + me) * 48 + q];
      ss = fmaf(v, v, ss);
    }
    sc[me] = rsqrtf(ss / 48.f + 1e-8f) * 0.14433756729740643f;  // /sqrt(48)
  }
  __syncthreads();

  v8f c = {0.f,0.f,0.f,0.f,0.f,0.f,0.f,0.f};
#pragma unroll
  for (int kc = 0; kc < 2; ++kc) {
    v16h aW = packA(A.W0, 48, me, kc * 32, 48, lane);
    v16h b;
    const int off = kc * 32 + ((lane & 16) ? 16 : 0);
#pragma unroll
    for (int j = 0; j < 16; ++j) {
      int v = off + j;
      b[j] = (v < 48) ? (_Float16)(A.acc2[(n0 + me) * 48 + v] * sc[me])
                      : (_Float16)0.f;
    }
    c = wmma16(aW, b, c);
  }

  const int n  = n0 + me;
  const int bb = A.batch[n];
#pragma unroll
  for (int g = 0; g < 8; ++g) {
    int u = g + half * 8;
    unsafeAtomicAdd(&A.pool[bb * 16 + u], sspf(c[g]));
  }
  if (half == 0) unsafeAtomicAdd(&A.cnt[bb], 1.f);
}

// ---------------------------------------------------------------------------
__global__ void k_head(const float* __restrict__ pool, const float* __restrict__ cnt,
                       const float* __restrict__ W1, const float* __restrict__ b1,
                       const float* __restrict__ W2, const float* __restrict__ b2,
                       float* __restrict__ out) {
  const int b = threadIdx.x;
  if (b >= NBATCH) return;
  float g[16];
  const float cn = fmaxf(cnt[b], 1.f);
#pragma unroll
  for (int u = 0; u < 16; ++u) g[u] = pool[b * 16 + u] / cn;
  float o = b2[0];
  for (int j = 0; j < 64; ++j) {
    float h = b1[j];
#pragma unroll
    for (int u = 0; u < 16; ++u) h = fmaf(g[u], W1[u * 64 + j], h);
    h = (h > 0.f) ? h : (expf(h) - 1.f);      // elu
    o = fmaf(h, W2[j], o);
  }
  out[b] = o;
}

// ---------------------------------------------------------------------------
extern "C" void kernel_launch(void* const* d_in, const int* in_sizes, int n_in,
                              void* d_out, int out_size, void* d_ws, size_t ws_size,
                              hipStream_t stream) {
  (void)in_sizes; (void)n_in; (void)out_size; (void)ws_size;

  // Input order = setup_inputs() dict + jax pytree (key-sorted) param leaves:
  //  0:x 1:edge_index 2:edge_attr 3:batch
  //  4..7:   fc  {W1, W2, b1, b2}
  //  8..19:  l1.convs  3 paths x {W1, W2, b1, b2}
  //  20:     l1.in_lin0
  //  21..23: l1.nl_bias[0..2]
  //  24..26: l1.out_lin[0..2]
  //  27..86: l2.convs  15 paths x 4 (sorted triple strings)
  //  87..89: l2.in_lin[0..2]   90..92: l2.nl_bias   93..95: l2.out_lin
  const float* x     = (const float*)d_in[0];
  const int*   ei    = (const int*)d_in[1];
  const float* ea    = (const float*)d_in[2];
  const int*   batch = (const int*)d_in[3];
  auto leaf = [&](int i) { return (const float*)d_in[i]; };

  float* ws    = (float*)d_ws;
  float* h0    = ws;                      // N*16
  float* ins0  = h0    + N_NODES * 16;    // N*16
  float* ins1  = ins0  + N_NODES * 16;    // N*48
  float* ins2  = ins1  + N_NODES * 48;    // N*80
  float* acc10 = ins2  + N_NODES * 80;    // N*16
  float* acc11 = acc10 + N_NODES * 16;    // N*48
  float* acc12 = acc11 + N_NODES * 48;    // N*80
  float* acc2  = acc12 + N_NODES * 80;    // N*48
  float* pool  = acc2  + N_NODES * 48;    // 64*16
  float* cnt   = pool  + NBATCH * 16;     // 64

  const int zn = N_NODES * (16 + 48 + 80 + 48) + NBATCH * 16 + NBATCH;
  k_zero<<<(zn + 255) / 256, 256, 0, stream>>>(acc10, zn);

  k_node_h0<<<N_NODES / 16, 32, 0, stream>>>(x, leaf(20), h0);

  EdgeArgs e1{};
  e1.ei = ei; e1.ea = ea;
  e1.inp[0] = e1.inp[1] = e1.inp[2] = h0;
  e1.outp[0] = acc10; e1.outp[1] = acc11; e1.outp[2] = acc12;
  for (int o = 0; o < 3; ++o) {
    int b = 8 + 4 * o;
    e1.rad[o] = Radial{leaf(b + 0), leaf(b + 2), leaf(b + 1), leaf(b + 3)};
  }
  k_edge<1><<<N_EDGES / (16 * EG), 32, 0, stream>>>(e1);

  L1PostArgs lp{};
  lp.acc[0] = acc10; lp.acc[1] = acc11; lp.acc[2] = acc12;
  lp.ins[0] = ins0;  lp.ins[1] = ins1;  lp.ins[2] = ins2;
  for (int o = 0; o < 3; ++o) {
    lp.outW[o] = leaf(24 + o);
    lp.bias[o] = leaf(21 + o);
    lp.inW2[o] = leaf(87 + o);
  }
  k_l1post<<<N_NODES / 16, 32, 0, stream>>>(lp);

  EdgeArgs e2{};
  e2.ei = ei; e2.ea = ea;
  e2.inp[0] = ins0; e2.inp[1] = ins1; e2.inp[2] = ins2;
  e2.outp[0] = e2.outp[1] = e2.outp[2] = acc2;
  const int pbase[3] = {27 + 4 * 0, 27 + 4 * 4, 27 + 4 * 12};  // (0,0,0),(1,1,0),(2,2,0)
  for (int p = 0; p < 3; ++p) {
    int b = pbase[p];
    e2.rad[p] = Radial{leaf(b + 0), leaf(b + 2), leaf(b + 1), leaf(b + 3)};
  }
  k_edge<2><<<N_EDGES / (16 * EG), 32, 0, stream>>>(e2);

  L2PostArgs l2a{};
  l2a.acc2 = acc2; l2a.W0 = leaf(93); l2a.batch = batch;
  l2a.pool = pool; l2a.cnt = cnt;
  k_l2post<<<N_NODES / 16, 32, 0, stream>>>(l2a);

  k_head<<<1, 64, 0, stream>>>(pool, cnt, leaf(4), leaf(6), leaf(5), leaf(7),
                               (float*)d_out);
}
